// BlockMoVaE_64372969832748
// MI455X (gfx1250) — matrix operand order repaired and checked by reading, not compiled, not run on hardware
//
#include <hip/hip_runtime.h>

// ---------------------------------------------------------------- types
typedef __bf16 bf16;
typedef bf16 v16bf __attribute__((ext_vector_type(16)));
typedef float v8f __attribute__((ext_vector_type(8)));
typedef unsigned int v4u __attribute__((ext_vector_type(4)));
typedef unsigned int v4ui __attribute__((ext_vector_type(4)));
typedef int v4i __attribute__((ext_vector_type(4)));
typedef int v8i __attribute__((ext_vector_type(8)));

union Frag32 { v4u u[2]; v16bf v; };

#define WMMA_BF16(a, b, c) \
  __builtin_amdgcn_wmma_f32_16x16x32_bf16(false, (a), false, (b), (short)0, (c), false, false)

// LDS byte offset of a generic pointer to __shared__ memory:
// flat shared addresses map to LDS by truncation to [31:0] (ISA 10.2 aperture rules).
__device__ __forceinline__ unsigned lds_off_of(const void* p) {
  return (unsigned)(unsigned long long)p;
}

// per-lane async 16B global->LDS copy (ASYNCcnt)
__device__ __forceinline__ void async_load_b128(unsigned lds_off, const void* gptr) {
  unsigned long long ga = (unsigned long long)gptr;
  asm volatile("global_load_async_to_lds_b128 %0, %1, off" ::"v"(lds_off), "v"(ga)
               : "memory");
}

__device__ __forceinline__ void wait_asynccnt0() {
  asm volatile("s_wait_asynccnt 0" ::: "memory");
}

// ---------------------------------------------------------------- model dims (fixed by reference)
constexpr int cB   = 2;
constexpr int cT   = 2048;
constexpr int cC   = 1024;
constexpr int cNH  = 16;
constexpr int cNKV = 8;
constexpr int cHD  = 64;
constexpr int cE   = 8;      // MLP experts
constexpr int cHID = 2048;
constexpr int cVOC = 50257;
constexpr int cN   = cB * cT;        // 4096 tokens
constexpr int cKV  = cNKV * cHD;     // 512
constexpr int cEt  = 10;             // total routed experts (8 MLP + 2 VE)

// ---------------------------------------------------------------- rmsnorm (f32 in -> bf16 out)
__global__ __launch_bounds__(256) void k_rmsnorm(const float* __restrict__ in,
                                                 bf16* __restrict__ out, int C) {
  __shared__ float red[9];
  int row = blockIdx.x;
  const float* p = in + (size_t)row * C;
  float ss = 0.f;
  for (int c = threadIdx.x; c < C; c += 256) { float v = p[c]; ss += v * v; }
#pragma unroll
  for (int d = 16; d >= 1; d >>= 1) ss += __shfl_xor(ss, d, 32);
  if ((threadIdx.x & 31) == 0) red[threadIdx.x >> 5] = ss;
  __syncthreads();
  if (threadIdx.x == 0) {
    float t = 0.f;
    for (int i = 0; i < 8; i++) t += red[i];
    red[8] = rsqrtf(t / (float)C + 1e-6f);
  }
  __syncthreads();
  float sc = red[8];
  for (int c = threadIdx.x; c < C; c += 256) out[(size_t)row * C + c] = (bf16)(p[c] * sc);
}

// ---------------------------------------------------------------- weight transpose: f32 [R][S] -> bf16 [S][R]
__global__ __launch_bounds__(256) void k_transpose(const float* __restrict__ in,
                                                   bf16* __restrict__ out, int R, int S) {
  __shared__ float tile[32][33];
  int bx = blockIdx.x * 32;  // S
  int by = blockIdx.y * 32;  // R
  int tx = threadIdx.x, ty = threadIdx.y;  // (32,8)
#pragma unroll
  for (int i = 0; i < 32; i += 8)
    tile[ty + i][tx] = in[(size_t)(by + ty + i) * S + bx + tx];
  __syncthreads();
#pragma unroll
  for (int i = 0; i < 32; i += 8)
    out[(size_t)(bx + ty + i) * R + by + tx] = (bf16)tile[tx][ty + i];
}

// ---------------------------------------------------------------- WMMA GEMM
//   C[M,N] = A[M,K] (bf16, row-major) x Bt[K,N] (bf16, = W^T)
//   block = 256 threads (8 waves), block tile 128x64, wave tile 32x32 (2x2 WMMA)
//   A tile staged with per-lane GLOBAL_LOAD_ASYNC_TO_LDS_B128 (ASYNCcnt)
//   B tile staged with one TDM tensor_load_to_lds per K-step      (TENSORcnt)
//   mode 0: outf = acc
//   mode 1: outf = acc + resid                 (wo projection + residual)
//   mode 2: outb = bf16(relu(acc)^2)           (MoE up)
//   mode 3: outf += gates[row*10+expert]*acc   (MoE down, gated accumulate)
#define GBM 128
#define GBN 64
#define GBK 32
__global__ __launch_bounds__(256) void k_gemm(const bf16* __restrict__ A,
                                              const bf16* __restrict__ Bt,
                                              int M, int Nn, int K, int mode,
                                              float* __restrict__ outf,
                                              bf16* __restrict__ outb,
                                              const float* __restrict__ resid,
                                              const float* __restrict__ gates, int expert) {
  __shared__ alignas(32) bf16 As[GBM][GBK + 8];  // row stride 40 elems (80B)
  __shared__ alignas(32) bf16 Bs[GBK][GBN];      // row stride 64 elems (128B), TDM-packed
  int tid = threadIdx.x;
  int lane = tid & 31, wid = tid >> 5;
  int half = lane >> 4, l16 = lane & 15;
  int m0 = blockIdx.x * GBM, n0 = blockIdx.y * GBN;
  int wm = (wid >> 1) * 32, wn = (wid & 1) * 32;

  v8f acc00 = {}, acc01 = {}, acc10 = {}, acc11 = {};

  int idxA = tid * 2;
  int rowA = idxA >> 2, segA = idxA & 3;  // A tile 128x32, 2x16B per thread
  unsigned ldsA0 = lds_off_of(&As[rowA][segA * 8]);
  unsigned ldsA1 = lds_off_of(&As[rowA][segA * 8 + 8]);
  unsigned ldsB = lds_off_of(&Bs[0][0]);

  for (int k0 = 0; k0 < K; k0 += GBK) {
    // ---- A tile: async per-lane copies, bypassing VGPRs
    const bf16* gA = A + (size_t)(m0 + rowA) * K + k0 + segA * 8;
    async_load_b128(ldsA0, gA);
    async_load_b128(ldsA1, gA + 8);

    // ---- B tile: Tensor Data Mover, one 32x64 bf16 tile (issued by wave 0)
    if (wid == 0) {
      unsigned long long gB = (unsigned long long)(Bt + (size_t)k0 * Nn + n0);
      v4ui g0;
      g0.x = 1u;                     // count=1, user descriptor
      g0.y = ldsB;                   // lds_addr (bytes)
      g0.z = (unsigned)gB;           // global_addr[31:0]
      g0.w = (unsigned)((gB >> 32) & 0x1ffffffu) | 0x80000000u;  // addr[56:32] | type=2
      v8i g1;
      g1[0] = 0x10000;               // workgroup_mask=0, data_size=1 (2 bytes)
      g1[1] = (Nn & 0xffff) << 16;   // tensor_dim0[15:0]  (bits 79:48)
      g1[2] = (int)(((unsigned)Nn >> 16) | ((unsigned)(K & 0xffff) << 16));  // dim0 hi | dim1 lo
      g1[3] = (int)(((unsigned)K >> 16) | ((unsigned)GBN << 16));            // dim1 hi | tile_dim0
      g1[4] = GBK;                   // tile_dim1 = 32 rows
      g1[5] = Nn;                    // tensor_dim0_stride[31:0] (elements)
      g1[6] = 0;
      g1[7] = 0;
      v4i gz = {0, 0, 0, 0};
      v8i gz8 = {0, 0, 0, 0, 0, 0, 0, 0};
      __builtin_amdgcn_tensor_load_to_lds(g0, g1, gz, gz, gz8, 0);
    }

    if (k0 + GBK < K) {
      __builtin_prefetch(A + (size_t)(m0 + rowA) * K + k0 + GBK + segA * 8, 0, 1);
      __builtin_prefetch(Bt + (size_t)(k0 + GBK) * Nn + n0, 0, 1);
    }

    wait_asynccnt0();
    __builtin_amdgcn_s_wait_tensorcnt(0);
    __syncthreads();

    Frag32 fa0, fa1;
    fa0.u[0] = *(const v4u*)&As[wm + l16][half * 8];
    fa0.u[1] = *(const v4u*)&As[wm + l16][half * 8 + 16];
    fa1.u[0] = *(const v4u*)&As[wm + 16 + l16][half * 8];
    fa1.u[1] = *(const v4u*)&As[wm + 16 + l16][half * 8 + 16];
    v16bf fb0, fb1;
#pragma unroll
    for (int s = 0; s < 16; s++) {
      int kk = (s < 8) ? half * 8 + s : 16 + half * 8 + (s - 8);
      fb0[s] = Bs[kk][wn + l16];
      fb1[s] = Bs[kk][wn + 16 + l16];
    }

    acc00 = WMMA_BF16(fa0.v, fb0, acc00);
    acc01 = WMMA_BF16(fa0.v, fb1, acc01);
    acc10 = WMMA_BF16(fa1.v, fb0, acc10);
    acc11 = WMMA_BF16(fa1.v, fb1, acc11);
    __syncthreads();
  }

  v8f accs[2][2] = {{acc00, acc01}, {acc10, acc11}};
#pragma unroll
  for (int i = 0; i < 2; i++)
#pragma unroll
    for (int j = 0; j < 2; j++)
#pragma unroll
      for (int r = 0; r < 8; r++) {
        int row = m0 + wm + i * 16 + half * 8 + r;
        int col = n0 + wn + j * 16 + l16;
        float v = accs[i][j][r];
        size_t off = (size_t)row * Nn + col;
        if (mode == 0) {
          outf[off] = v;
        } else if (mode == 1) {
          outf[off] = v + resid[off];
        } else if (mode == 2) {
          float t = v > 0.f ? v : 0.f;
          outb[off] = (bf16)(t * t);
        } else {
          float g = gates[(size_t)row * cEt + expert];
          outf[off] += g * v;
        }
      }
}

// ---------------------------------------------------------------- RoPE + per-head rmsnorm -> bf16
// one wave per (token, head); lane handles rotation pair (d, d+32)
__global__ void k_rope(const float* __restrict__ in, const float* __restrict__ cosb,
                       const float* __restrict__ sinb, bf16* __restrict__ out_plain,
                       bf16* __restrict__ out_tr, int heads, int Tdim, int is_k) {
  int tid = threadIdx.x;
  int lane = tid & 31, h = tid >> 5;
  int row = blockIdx.x;  // b*T + t
  int t = row & (Tdim - 1);
  int b = row / Tdim;
  int W = heads * cHD;
  size_t base = (size_t)row * W + h * cHD;
  float x1 = in[base + lane];
  float x2 = in[base + 32 + lane];
  float c = cosb[t * 32 + lane];
  float s = sinb[t * 32 + lane];
  float o1 = x1 * c + x2 * s;
  float o2 = -x1 * s + x2 * c;
  float ss = o1 * o1 + o2 * o2;
#pragma unroll
  for (int d = 16; d >= 1; d >>= 1) ss += __shfl_xor(ss, d, 32);
  float sc = rsqrtf(ss / (float)cHD + 1e-6f);
  o1 *= sc; o2 *= sc;
  if (!is_k) {
    out_plain[base + lane] = (bf16)o1;
    out_plain[base + 32 + lane] = (bf16)o2;
  } else {
    size_t tb = ((size_t)b * heads + h) * cHD;  // [B][heads][HD][T]
    out_tr[(tb + lane) * Tdim + t] = (bf16)o1;
    out_tr[(tb + 32 + lane) * Tdim + t] = (bf16)o2;
  }
}

// ---------------------------------------------------------------- f32 -> bf16 convert
__global__ void k_cvt(const float* __restrict__ in, bf16* __restrict__ out, int n) {
  int i = blockIdx.x * 256 + threadIdx.x;
  if (i < n) out[i] = (bf16)in[i];
}

// ---------------------------------------------------------------- flash attention (WMMA, online softmax)
// grid (T/128, B*NH), block 256 (8 waves); each wave owns 16 query rows.
__global__ __launch_bounds__(256) void k_attn(const bf16* __restrict__ qb,
                                              const bf16* __restrict__ kt,
                                              const bf16* __restrict__ vb,
                                              bf16* __restrict__ yb, int Tdim) {
  __shared__ alignas(32) bf16 P[8][16][40];  // per-wave P-transpose scratch
  int tid = threadIdx.x;
  int lane = tid & 31, wid = tid >> 5;
  int half = lane >> 4, l16 = lane & 15;
  int bh = blockIdx.y;
  int b = bh >> 4;        // NH = 16
  int h = bh & 15;
  int kvh = h >> 1;       // GQA rep = 2
  int q0 = (blockIdx.x * 8 + wid) * 16;

  const bf16* Qp = qb + (size_t)b * Tdim * cC;
  const bf16* Kp = kt + ((size_t)b * cNKV + kvh) * cHD * Tdim;  // [d][T]
  const bf16* Vp = vb + (size_t)b * Tdim * cKV + kvh * cHD;

  // Q A-fragments (16x64 -> two K=32 chunks)
  Frag32 aq0, aq1;
  {
    const bf16* s0 = Qp + (size_t)(q0 + l16) * cC + h * cHD + half * 8;
    aq0.u[0] = *(const v4u*)s0;
    aq0.u[1] = *(const v4u*)(s0 + 16);
    const bf16* s1 = s0 + 32;
    aq1.u[0] = *(const v4u*)s1;
    aq1.u[1] = *(const v4u*)(s1 + 16);
  }

  v8f o0 = {}, o1 = {}, o2 = {}, o3 = {};
  float mrow[8], lrow[8];
#pragma unroll
  for (int r = 0; r < 8; r++) { mrow[r] = -3e38f; lrow[r] = 0.f; }

  int jend = q0 + 15;
  for (int j0 = 0; j0 <= jend; j0 += 32) {
    // S = Q K^T for 16 x 32 key block
    v8f s0 = {}, s1 = {};
    {
      v16bf b0, b1, b2, b3;
#pragma unroll
      for (int s = 0; s < 16; s++) {
        int kk = (s < 8) ? half * 8 + s : 16 + half * 8 + (s - 8);
        b0[s] = Kp[(size_t)kk * Tdim + j0 + l16];
        b1[s] = Kp[(size_t)kk * Tdim + j0 + 16 + l16];
        b2[s] = Kp[(size_t)(32 + kk) * Tdim + j0 + l16];
        b3[s] = Kp[(size_t)(32 + kk) * Tdim + j0 + 16 + l16];
      }
      s0 = WMMA_BF16(aq0.v, b0, s0);
      s1 = WMMA_BF16(aq0.v, b1, s1);
      s0 = WMMA_BF16(aq1.v, b2, s0);
      s1 = WMMA_BF16(aq1.v, b3, s1);
    }
    // mask + online softmax (half-wave row reductions)
#pragma unroll
    for (int r = 0; r < 8; r++) {
      int rowi = q0 + half * 8 + r;
      int c0 = j0 + l16, c1 = j0 + 16 + l16;
      float v0 = (c0 <= rowi) ? s0[r] * 0.125f : -3e38f;
      float v1 = (c1 <= rowi) ? s1[r] * 0.125f : -3e38f;
      float vmax = fmaxf(v0, v1);
#pragma unroll
      for (int d = 1; d < 16; d <<= 1) vmax = fmaxf(vmax, __shfl_xor(vmax, d, 32));
      float mnew = fmaxf(mrow[r], vmax);
      float cf = __expf(mrow[r] - mnew);
      float p0 = __expf(v0 - mnew);
      float p1 = __expf(v1 - mnew);
      float ps = p0 + p1;
#pragma unroll
      for (int d = 1; d < 16; d <<= 1) ps += __shfl_xor(ps, d, 32);
      lrow[r] = lrow[r] * cf + ps;
      mrow[r] = mnew;
      o0[r] *= cf; o1[r] *= cf; o2[r] *= cf; o3[r] *= cf;
      P[wid][half * 8 + r][l16] = (bf16)p0;
      P[wid][half * 8 + r][16 + l16] = (bf16)p1;
    }
    // P (16x32) as A-fragment via LDS transpose
    Frag32 ap;
    ap.u[0] = *(const v4u*)&P[wid][l16][half * 8];
    ap.u[1] = *(const v4u*)&P[wid][l16][half * 8 + 16];
    // O += P V  (V block 32x64)
    v16bf bv0, bv1, bv2, bv3;
#pragma unroll
    for (int s = 0; s < 16; s++) {
      int kk = (s < 8) ? half * 8 + s : 16 + half * 8 + (s - 8);
      const bf16* vp = Vp + (size_t)(j0 + kk) * cKV + l16;
      bv0[s] = vp[0];
      bv1[s] = vp[16];
      bv2[s] = vp[32];
      bv3[s] = vp[48];
    }
    o0 = WMMA_BF16(ap.v, bv0, o0);
    o1 = WMMA_BF16(ap.v, bv1, o1);
    o2 = WMMA_BF16(ap.v, bv2, o2);
    o3 = WMMA_BF16(ap.v, bv3, o3);
  }

  bf16* yo = yb + (size_t)b * Tdim * cC + h * cHD;
#pragma unroll
  for (int r = 0; r < 8; r++) {
    float inv = 1.0f / lrow[r];
    bf16* dst = yo + (size_t)(q0 + half * 8 + r) * cC;
    dst[l16] = (bf16)(o0[r] * inv);
    dst[16 + l16] = (bf16)(o1[r] * inv);
    dst[32 + l16] = (bf16)(o2[r] * inv);
    dst[48 + l16] = (bf16)(o3[r] * inv);
  }
}

// ---------------------------------------------------------------- router logits: one wave per (token, expert)
__global__ void k_router(const bf16* __restrict__ xf, const float* __restrict__ rw,
                         float* __restrict__ logits) {
  int n = blockIdx.x;
  int e = threadIdx.y;   // 10 experts
  int lane = threadIdx.x;
  const bf16* xp = xf + (size_t)n * cC;
  const float* wp = rw + (size_t)e * cC;
  float acc = 0.f;
  for (int k = lane; k < cC; k += 32) acc += (float)xp[k] * wp[k];
#pragma unroll
  for (int d = 16; d >= 1; d >>= 1) acc += __shfl_xor(acc, d, 32);
  if (lane == 0) logits[(size_t)n * cEt + e] = acc;
}

// ---------------------------------------------------------------- sigmoid + top-2 + normalize
__global__ void k_topk(const float* __restrict__ logits, const float* __restrict__ bias,
                       float* __restrict__ gates) {
  int n = blockIdx.x * 256 + threadIdx.x;
  if (n >= cN) return;
  float sig[cEt], sel[cEt];
#pragma unroll
  for (int e = 0; e < cEt; e++) {
    float l = logits[(size_t)n * cEt + e];
    float s = 1.0f / (1.0f + __expf(-l));
    sig[e] = s;
    sel[e] = s + bias[e];
  }
  int i0 = 0;
#pragma unroll
  for (int e = 1; e < cEt; e++) if (sel[e] > sel[i0]) i0 = e;
  int i1 = (i0 == 0) ? 1 : 0;
#pragma unroll
  for (int e = 0; e < cEt; e++) if (e != i0 && sel[e] > sel[i1]) i1 = e;
  float w0 = sig[i0], w1 = sig[i1];
  float inv = 1.0f / (w0 + w1);
#pragma unroll
  for (int e = 0; e < cEt; e++) gates[(size_t)n * cEt + e] = 0.f;
  gates[(size_t)n * cEt + i0] = w0 * inv;
  gates[(size_t)n * cEt + i1] = w1 * inv;
}

// ---------------------------------------------------------------- out = x2 + gated value-embedding experts
__global__ void k_ve(const float* __restrict__ x2, const float* __restrict__ gates,
                     const int* __restrict__ tok, const float* __restrict__ ve,
                     float* __restrict__ out) {
  int idx = blockIdx.x * 256 + threadIdx.x;
  int n = idx >> 10;     // C = 1024
  int c = idx & 1023;
  float g8 = gates[(size_t)n * cEt + 8];
  float g9 = gates[(size_t)n * cEt + 9];
  int t = tok[n];
  float v = x2[idx];
  v += g8 * ve[(size_t)t * cC + c];
  v += g9 * ve[((size_t)cVOC + t) * cC + c];
  out[idx] = v;
}

// ---------------------------------------------------------------- launch
extern "C" void kernel_launch(void* const* d_in, const int* in_sizes, int n_in,
                              void* d_out, int out_size, void* d_ws, size_t ws_size,
                              hipStream_t stream) {
  (void)in_sizes; (void)n_in; (void)out_size; (void)ws_size;
  const float* x    = (const float*)d_in[0];
  const int*   tok  = (const int*)d_in[1];
  const float* cosb = (const float*)d_in[2];
  const float* sinb = (const float*)d_in[3];
  // d_in[4] = window_size == T -> mask reduces to pure causal
  const float* wq   = (const float*)d_in[5];
  const float* wk   = (const float*)d_in[6];
  const float* wv   = (const float*)d_in[7];
  const float* wo   = (const float*)d_in[8];
  const float* wup  = (const float*)d_in[9];
  const float* wdn  = (const float*)d_in[10];
  const float* rw   = (const float*)d_in[11];
  const float* rb   = (const float*)d_in[12];
  const float* ve   = (const float*)d_in[13];
  float* out = (float*)d_out;

  char* wsb = (char*)d_ws;
  size_t off = 0;
  auto alloc = [&](size_t bytes) -> char* {
    char* p = wsb + off;
    off += (bytes + 255) & ~(size_t)255;
    return p;
  };
  bf16*  xn    = (bf16*)alloc((size_t)cN * cC * 2);
  float* qf    = (float*)alloc((size_t)cN * cC * 4);
  float* kf    = (float*)alloc((size_t)cN * cKV * 4);
  float* vf    = (float*)alloc((size_t)cN * cKV * 4);
  bf16*  qb    = (bf16*)alloc((size_t)cN * cC * 2);
  bf16*  ktb   = (bf16*)alloc((size_t)cN * cKV * 2);
  bf16*  vb    = (bf16*)alloc((size_t)cN * cKV * 2);
  bf16*  yb    = (bf16*)alloc((size_t)cN * cC * 2);
  float* x2    = (float*)alloc((size_t)cN * cC * 4);
  bf16*  xfb   = (bf16*)alloc((size_t)cN * cC * 2);
  float* logit = (float*)alloc((size_t)cN * cEt * 4);
  float* gates = (float*)alloc((size_t)cN * cEt * 4);
  bf16*  hid   = (bf16*)alloc((size_t)cN * cHID * 2);
  bf16*  wqt   = (bf16*)alloc((size_t)cC * cC * 2);
  bf16*  wkt   = (bf16*)alloc((size_t)cC * cKV * 2);
  bf16*  wvt   = (bf16*)alloc((size_t)cC * cKV * 2);
  bf16*  wot   = (bf16*)alloc((size_t)cC * cC * 2);
  bf16*  wupt  = (bf16*)alloc((size_t)cE * cC * cHID * 2);
  bf16*  wdnt  = (bf16*)alloc((size_t)cE * cHID * cC * 2);

  dim3 tb(32, 8);
  // weight bf16 transposes: W[N,K] -> Wt[K,N]
  k_transpose<<<dim3(cC / 32, cC / 32), tb, 0, stream>>>(wq, wqt, cC, cC);
  k_transpose<<<dim3(cC / 32, cKV / 32), tb, 0, stream>>>(wk, wkt, cKV, cC);
  k_transpose<<<dim3(cC / 32, cKV / 32), tb, 0, stream>>>(wv, wvt, cKV, cC);
  k_transpose<<<dim3(cC / 32, cC / 32), tb, 0, stream>>>(wo, wot, cC, cC);
  for (int e = 0; e < cE; e++) {
    k_transpose<<<dim3(cC / 32, cHID / 32), tb, 0, stream>>>(
        wup + (size_t)e * cHID * cC, wupt + (size_t)e * cC * cHID, cHID, cC);
    k_transpose<<<dim3(cHID / 32, cC / 32), tb, 0, stream>>>(
        wdn + (size_t)e * cC * cHID, wdnt + (size_t)e * cHID * cC, cC, cHID);
  }

  // 1. xn = rmsnorm(x)
  k_rmsnorm<<<cN, 256, 0, stream>>>(x, xn, cC);

  // 2. q/k/v projections
  k_gemm<<<dim3(cN / GBM, cC / GBN), 256, 0, stream>>>(xn, wqt, cN, cC, cC, 0, qf, nullptr, nullptr, nullptr, 0);
  k_gemm<<<dim3(cN / GBM, cKV / GBN), 256, 0, stream>>>(xn, wkt, cN, cKV, cC, 0, kf, nullptr, nullptr, nullptr, 0);
  k_gemm<<<dim3(cN / GBM, cKV / GBN), 256, 0, stream>>>(xn, wvt, cN, cKV, cC, 0, vf, nullptr, nullptr, nullptr, 0);

  // 3. rope + head rmsnorm; K stored transposed [b][kv][d][T], V converted to bf16
  k_rope<<<cN, cNH * 32, 0, stream>>>(qf, cosb, sinb, qb, nullptr, cNH, cT, 0);
  k_rope<<<cN, cNKV * 32, 0, stream>>>(kf, cosb, sinb, nullptr, ktb, cNKV, cT, 1);
  k_cvt<<<(cN * cKV) / 256, 256, 0, stream>>>(vf, vb, cN * cKV);

  // 4. flash attention
  k_attn<<<dim3(cT / 128, cB * cNH), 256, 0, stream>>>(qb, ktb, vb, yb, cT);

  // 5. x2 = x + y @ wo^T
  k_gemm<<<dim3(cN / GBM, cC / GBN), 256, 0, stream>>>(yb, wot, cN, cC, cC, 1, x2, nullptr, x, nullptr, 0);

  // 6. xf = rmsnorm(x2); router
  k_rmsnorm<<<cN, 256, 0, stream>>>(x2, xfb, cC);
  k_router<<<cN, dim3(32, cEt), 0, stream>>>(xfb, rw, logit);
  k_topk<<<cN / 256, 256, 0, stream>>>(logit, rb, gates);

  // 7. out = x2 + gated value embeddings
  k_ve<<<(cN * cC) / 256, 256, 0, stream>>>(x2, gates, tok, ve, out);

  // 8. MoE experts: hidden = relu(xf @ wup^T)^2 ; out += gate * (hidden @ wdn^T)
  for (int e = 0; e < cE; e++) {
    k_gemm<<<dim3(cN / GBM, cHID / GBN), 256, 0, stream>>>(
        xfb, wupt + (size_t)e * cC * cHID, cN, cHID, cC, 2, nullptr, hid, nullptr, nullptr, 0);
    k_gemm<<<dim3(cN / GBM, cC / GBN), 256, 0, stream>>>(
        hid, wdnt + (size_t)e * cHID * cC, cN, cC, cHID, 3, out, nullptr, nullptr, gates, e);
  }
}
